// MarineDebrisGNN_89953795047697
// MI455X (gfx1250) — matrix operand order repaired and checked by reading, not compiled, not run on hardware
//
#include <hip/hip_runtime.h>
#include <hip/hip_bf16.h>
#include <math.h>

#define NN   50000
#define NP   50016          /* padded to multiple of 16 */
#define EEDG 800000
#define EE2  (EEDG + NN)    /* edges + self loops */
#define TT   168

typedef __attribute__((ext_vector_type(16))) _Float16 v16h;
typedef __attribute__((ext_vector_type(8)))  _Float16 v8h;
typedef __attribute__((ext_vector_type(8)))  float    v8f;

__device__ __forceinline__ float sigm(float x) { return 1.0f / (1.0f + expf(-x)); }
__device__ __forceinline__ unsigned fkey(float f) {
  unsigned u = __float_as_uint(f);
  return (u & 0x80000000u) ? ~u : (u | 0x80000000u);
}
__device__ __forceinline__ float funkey(unsigned k) {
  unsigned u = (k & 0x80000000u) ? (k ^ 0x80000000u) : ~k;
  return __uint_as_float(u);
}

/* ---------------- generic zero fill ---------------- */
__global__ void k_zero(float* p, int n) {
  int i = blockIdx.x * 256 + threadIdx.x;
  if (i < n) p[i] = 0.0f;
}

/* ---------------- LSTM (2 layers) + embedding, one block of 64 ---------------- */
__global__ void k_lstm(const float* __restrict__ env,
                       const float* __restrict__ Wih0, const float* __restrict__ Whh0,
                       const float* __restrict__ bih0, const float* __restrict__ bhh0,
                       const float* __restrict__ Wih1, const float* __restrict__ Whh1,
                       const float* __restrict__ bih1, const float* __restrict__ bhh1,
                       const float* __restrict__ embW, const float* __restrict__ embB,
                       float* __restrict__ hs0, float* __restrict__ emb_out) {
  __shared__ float h[64], c[64];
  int j = threadIdx.x;
  h[j] = 0.0f; c[j] = 0.0f;
  __syncthreads();
  /* layer 0: input dim 8 */
  for (int t = 0; t < TT; ++t) {
    float gi = bih0[j] + bhh0[j];
    float gf = bih0[64 + j] + bhh0[64 + j];
    float gg = bih0[128 + j] + bhh0[128 + j];
    float go = bih0[192 + j] + bhh0[192 + j];
    for (int k = 0; k < 8; ++k) {
      float xv = env[t * 8 + k];
      gi += Wih0[j * 8 + k] * xv;
      gf += Wih0[(64 + j) * 8 + k] * xv;
      gg += Wih0[(128 + j) * 8 + k] * xv;
      go += Wih0[(192 + j) * 8 + k] * xv;
    }
    for (int k = 0; k < 64; ++k) {
      float hv = h[k];
      gi += Whh0[j * 64 + k] * hv;
      gf += Whh0[(64 + j) * 64 + k] * hv;
      gg += Whh0[(128 + j) * 64 + k] * hv;
      go += Whh0[(192 + j) * 64 + k] * hv;
    }
    float cn = sigm(gf) * c[j] + sigm(gi) * tanhf(gg);
    float hn = sigm(go) * tanhf(cn);
    __syncthreads();
    c[j] = cn; h[j] = hn; hs0[t * 64 + j] = hn;
    __syncthreads();
  }
  /* reset state for layer 1 */
  h[j] = 0.0f; c[j] = 0.0f;
  __syncthreads();
  for (int t = 0; t < TT; ++t) {
    float gi = bih1[j] + bhh1[j];
    float gf = bih1[64 + j] + bhh1[64 + j];
    float gg = bih1[128 + j] + bhh1[128 + j];
    float go = bih1[192 + j] + bhh1[192 + j];
    for (int k = 0; k < 64; ++k) {
      float xv = hs0[t * 64 + k];
      gi += Wih1[j * 64 + k] * xv;
      gf += Wih1[(64 + j) * 64 + k] * xv;
      gg += Wih1[(128 + j) * 64 + k] * xv;
      go += Wih1[(192 + j) * 64 + k] * xv;
    }
    for (int k = 0; k < 64; ++k) {
      float hv = h[k];
      gi += Whh1[j * 64 + k] * hv;
      gf += Whh1[(64 + j) * 64 + k] * hv;
      gg += Whh1[(128 + j) * 64 + k] * hv;
      go += Whh1[(192 + j) * 64 + k] * hv;
    }
    float cn = sigm(gf) * c[j] + sigm(gi) * tanhf(gg);
    float hn = sigm(go) * tanhf(cn);
    __syncthreads();
    c[j] = cn; h[j] = hn;
    __syncthreads();
  }
  if (j < 32) {
    float acc = embB[j];
    for (int k = 0; k < 64; ++k) acc += embW[j * 64 + k] * h[k];
    emb_out[j] = acc;
  }
}

/* ---------------- weight prep: f16 B-transposed weights + edge-attn vectors ---------------- */
__global__ void k_prep_w(const float* __restrict__ g1W, const float* __restrict__ g2W,
                         const float* __restrict__ g1eW, const float* __restrict__ g1ae,
                         const float* __restrict__ g2eW, const float* __restrict__ g2ae,
                         _Float16* __restrict__ BT1, _Float16* __restrict__ BT2,
                         float* __restrict__ q1, float* __restrict__ q2) {
  int tid = threadIdx.x;
  /* BT1[n][k], n<256, k<64 (K padded 39->64) */
  for (int idx = tid; idx < 256 * 64; idx += 256) {
    int n = idx >> 6, k = idx & 63;
    BT1[idx] = (k < 39) ? (_Float16)g1W[k * 256 + n] : (_Float16)0.0f;
  }
  /* BT2[n][k], n<32, k<256 */
  for (int idx = tid; idx < 32 * 256; idx += 256) {
    int n = idx >> 8, k = idx & 255;
    BT2[idx] = (_Float16)g2W[k * 32 + n];
  }
  if (tid < 20) {
    int h = tid / 5, d = tid % 5;
    float acc = 0.0f;
    for (int cIt = 0; cIt < 64; ++cIt) acc += g1eW[d * 256 + h * 64 + cIt] * g1ae[h * 64 + cIt];
    q1[h * 5 + d] = acc;
  }
  if (tid < 5) {
    float acc = 0.0f;
    for (int cIt = 0; cIt < 32; ++cIt) acc += g2eW[tid * 32 + cIt] * g2ae[cIt];
    q2[tid] = acc;
  }
}

/* ---------------- build Xin f16 [NP,64]: x(7) | emb(32) | pad ---------------- */
__global__ void k_xin(const float* __restrict__ x, const float* __restrict__ emb,
                      _Float16* __restrict__ Xin) {
  int n = blockIdx.x * 256 + threadIdx.x;
  if (n >= NP) return;
  _Float16* row = Xin + (size_t)n * 64;
  if (n < NN) {
    for (int k = 0; k < 7; ++k)  row[k]     = (_Float16)x[n * 7 + k];
    for (int k = 0; k < 32; ++k) row[7 + k] = (_Float16)emb[k];
    for (int k = 39; k < 64; ++k) row[k] = (_Float16)0.0f;
  } else {
    for (int k = 0; k < 64; ++k) row[k] = (_Float16)0.0f;
  }
}

/* ---------------- degree + edge-attr sum per destination ---------------- */
__global__ void k_deg(const int* __restrict__ ei, const float* __restrict__ eattr,
                      float* __restrict__ deg, float* __restrict__ meane) {
  int e = blockIdx.x * 256 + threadIdx.x;
  if (e >= EEDG) return;
  int d = ei[EEDG + e];
  atomicAdd(&deg[d], 1.0f);
  for (int k = 0; k < 5; ++k) atomicAdd(&meane[d * 5 + k], eattr[e * 5 + k]);
}
__global__ void k_meandiv(const float* __restrict__ deg, float* __restrict__ meane) {
  int i = blockIdx.x * 256 + threadIdx.x;
  if (i >= NN * 5) return;
  float dg = deg[i / 5];
  meane[i] = meane[i] / fmaxf(dg, 1.0f);
}

/* ---------------- WMMA GEMM: C[M,ldc] = A[M,Kpad](f16) x B, B given K-major as BT[N,Kpad] ---- */
__global__ void k_gemm_wmma(const _Float16* __restrict__ A, const _Float16* __restrict__ BT,
                            float* __restrict__ C, int Mtiles, int Ntiles, int Kpad, int ldc) {
  int tile = blockIdx.x * 4 + (threadIdx.x >> 5);
  if (tile >= Mtiles * Ntiles) return;
  int mt = tile / Ntiles, nt = tile % Ntiles;
  int lane = threadIdx.x & 31;
  int hw = lane >> 4;         /* half-wave: 0 or 1 */
  int l15 = lane & 15;
  const _Float16* Arow = A + (size_t)(mt * 16 + l15) * Kpad + hw * 8;
  const _Float16* Brow = BT + (size_t)(nt * 16 + l15) * Kpad + hw * 8;
  v8f acc = {};
  for (int k0 = 0; k0 < Kpad; k0 += 32) {
    v8h a0 = *(const v8h*)(Arow + k0);       /* K = k0 + 8*hw + [0..7]        */
    v8h a1 = *(const v8h*)(Arow + k0 + 16);  /* K = k0 + 16 + 8*hw + [0..7]   */
    v8h b0 = *(const v8h*)(Brow + k0);       /* K = k0 + 8*hw + [0..7]        */
    v8h b1 = *(const v8h*)(Brow + k0 + 16);  /* K = k0 + 16 + 8*hw + [0..7]   */
    v16h a, b;
    for (int j = 0; j < 8; ++j) { a[j] = a0[j]; a[8 + j] = a1[j]; b[j] = b0[j]; b[8 + j] = b1[j]; }
    acc = __builtin_amdgcn_wmma_f32_16x16x32_f16(false, a, false, b, (short)0, acc, false, false);
  }
  /* C layout: lane half selects M block (0/8), VGPR v selects M row, lane&15 = N col */
  float* Cp = C + (size_t)(mt * 16 + hw * 8) * ldc + (nt * 16 + l15);
  for (int v = 0; v < 8; ++v) Cp[(size_t)v * ldc] = acc[v];
}

/* ---------------- per-node attention logits a_src/a_dst, GAT1 (4 heads x 64) ------------- */
__global__ void k_ad1(const float* __restrict__ H1, const float* __restrict__ asrc,
                      const float* __restrict__ adst, float* __restrict__ as1,
                      float* __restrict__ ad1) {
  int id = blockIdx.x * 256 + threadIdx.x;
  if (id >= NN * 4) return;
  int n = id >> 2, h = id & 3;
  const float* row = H1 + (size_t)n * 256 + h * 64;
  float s = 0.0f, d = 0.0f;
  for (int cIt = 0; cIt < 64; ++cIt) {
    float v = row[cIt];
    s += v * asrc[h * 64 + cIt];
    d += v * adst[h * 64 + cIt];
  }
  as1[id] = s; ad1[id] = d;
}

/* ---------------- GAT1: alpha + segment max ---------------- */
__global__ void k_alpha1(const int* __restrict__ ei, const float* __restrict__ eattr,
                         const float* __restrict__ meane, const float* __restrict__ as1,
                         const float* __restrict__ ad1, const float* __restrict__ q1,
                         float* __restrict__ alph, unsigned* __restrict__ amax) {
  int e = blockIdx.x * 256 + threadIdx.x;
  if (e >= EE2) return;
  int s, d; float eav[5];
  if (e < EEDG) {
    s = ei[e]; d = ei[EEDG + e];
    for (int k = 0; k < 5; ++k) eav[k] = eattr[e * 5 + k];
  } else {
    s = d = e - EEDG;
    for (int k = 0; k < 5; ++k) eav[k] = meane[(size_t)(e - EEDG) * 5 + k];
  }
  for (int h = 0; h < 4; ++h) {
    float ae = 0.0f;
    for (int k = 0; k < 5; ++k) ae += eav[k] * q1[h * 5 + k];
    float a = as1[s * 4 + h] + ad1[d * 4 + h] + ae;
    a = (a > 0.0f) ? a : 0.2f * a;
    alph[(size_t)e * 4 + h] = a;
    atomicMax(&amax[d * 4 + h], fkey(a));
  }
}

/* ---------------- GAT1: exp + segment sum ---------------- */
__global__ void k_exp1(const int* __restrict__ ei, const unsigned* __restrict__ amax,
                       float* __restrict__ alph, float* __restrict__ den) {
  int id = blockIdx.x * 256 + threadIdx.x;
  if (id >= EE2 * 4) return;
  int e = id >> 2, h = id & 3;
  int d = (e < EEDG) ? ei[EEDG + e] : (e - EEDG);
  float ex = expf(alph[id] - funkey(amax[d * 4 + h]));
  alph[id] = ex;
  atomicAdd(&den[d * 4 + h], ex);
}

/* ---------------- GAT1: weighted scatter-add of messages ---------------- */
__global__ void k_agg1(const int* __restrict__ ei, const float* __restrict__ alph,
                       const float* __restrict__ den, const float* __restrict__ H1,
                       float* __restrict__ out1) {
  int id = blockIdx.x * 256 + threadIdx.x;
  if (id >= EE2 * 4) return;
  int e = id >> 2, h = id & 3;
  int s = (e < EEDG) ? ei[e] : (e - EEDG);
  int d = (e < EEDG) ? ei[EEDG + e] : (e - EEDG);
  float att = alph[id] / (den[d * 4 + h] + 1e-16f);
  const float* srcv = H1 + (size_t)s * 256 + h * 64;
  float* dstv = out1 + (size_t)d * 256 + h * 64;
  __builtin_prefetch(srcv, 0, 3);
  for (int cIt = 0; cIt < 64; ++cIt) atomicAdd(&dstv[cIt], srcv[cIt] * att);
}

/* ---------------- ELU + f16 convert for GEMM2 input ---------------- */
__global__ void k_elu1(const float* __restrict__ out1, const float* __restrict__ bias,
                       _Float16* __restrict__ H1h) {
  int id = blockIdx.x * 256 + threadIdx.x;
  if (id >= NP * 256) return;
  int n = id >> 8, cIt = id & 255;
  float v = 0.0f;
  if (n < NN) {
    float t = out1[id] + bias[cIt];
    v = (t > 0.0f) ? t : (expf(t) - 1.0f);
  }
  H1h[id] = (_Float16)v;
}

/* ---------------- GAT2 (1 head, C=32) ---------------- */
__global__ void k_ad2(const float* __restrict__ H2, const float* __restrict__ asrc,
                      const float* __restrict__ adst, float* __restrict__ as2,
                      float* __restrict__ ad2) {
  int n = blockIdx.x * 256 + threadIdx.x;
  if (n >= NN) return;
  const float* row = H2 + (size_t)n * 32;
  float s = 0.0f, d = 0.0f;
  for (int cIt = 0; cIt < 32; ++cIt) { float v = row[cIt]; s += v * asrc[cIt]; d += v * adst[cIt]; }
  as2[n] = s; ad2[n] = d;
}
__global__ void k_alpha2(const int* __restrict__ ei, const float* __restrict__ eattr,
                         const float* __restrict__ meane, const float* __restrict__ as2,
                         const float* __restrict__ ad2, const float* __restrict__ q2,
                         float* __restrict__ alph, unsigned* __restrict__ amax) {
  int e = blockIdx.x * 256 + threadIdx.x;
  if (e >= EE2) return;
  int s, d; float ae = 0.0f;
  if (e < EEDG) {
    s = ei[e]; d = ei[EEDG + e];
    for (int k = 0; k < 5; ++k) ae += eattr[e * 5 + k] * q2[k];
  } else {
    s = d = e - EEDG;
    for (int k = 0; k < 5; ++k) ae += meane[(size_t)(e - EEDG) * 5 + k] * q2[k];
  }
  float a = as2[s] + ad2[d] + ae;
  a = (a > 0.0f) ? a : 0.2f * a;
  alph[e] = a;
  atomicMax(&amax[d], fkey(a));
}
__global__ void k_exp2(const int* __restrict__ ei, const unsigned* __restrict__ amax,
                       float* __restrict__ alph, float* __restrict__ den) {
  int e = blockIdx.x * 256 + threadIdx.x;
  if (e >= EE2) return;
  int d = (e < EEDG) ? ei[EEDG + e] : (e - EEDG);
  float ex = expf(alph[e] - funkey(amax[d]));
  alph[e] = ex;
  atomicAdd(&den[d], ex);
}
__global__ void k_agg2(const int* __restrict__ ei, const float* __restrict__ alph,
                       const float* __restrict__ den, const float* __restrict__ H2,
                       float* __restrict__ out2) {
  int e = blockIdx.x * 256 + threadIdx.x;
  if (e >= EE2) return;
  int s = (e < EEDG) ? ei[e] : (e - EEDG);
  int d = (e < EEDG) ? ei[EEDG + e] : (e - EEDG);
  float att = alph[e] / (den[d] + 1e-16f);
  const float* srcv = H2 + (size_t)s * 32;
  float* dstv = out2 + (size_t)d * 32;
  for (int cIt = 0; cIt < 32; ++cIt) atomicAdd(&dstv[cIt], srcv[cIt] * att);
}

/* ---------------- final: residual head + anchor ---------------- */
__global__ void k_final(const float* __restrict__ x, const float* __restrict__ scmax,
                        const float* __restrict__ out2, const float* __restrict__ g2b,
                        const float* __restrict__ outW, const float* __restrict__ outB,
                        float* __restrict__ y) {
  int n = blockIdx.x * 256 + threadIdx.x;
  if (n >= NN) return;
  float res = outB[0];
  const float* row = out2 + (size_t)n * 32;
  for (int cIt = 0; cIt < 32; ++cIt) res += (row[cIt] + g2b[cIt]) * outW[cIt];
  y[n] = x[n * 7 + 6] * scmax[0] + res;
}

extern "C" void kernel_launch(void* const* d_in, const int* in_sizes, int n_in,
                              void* d_out, int out_size, void* d_ws, size_t ws_size,
                              hipStream_t stream) {
  (void)in_sizes; (void)n_in; (void)out_size; (void)ws_size;
  const float* x     = (const float*)d_in[0];
  const int*   ei    = (const int*)  d_in[1];
  const float* eattr = (const float*)d_in[2];
  const float* scmax = (const float*)d_in[3];
  const float* env   = (const float*)d_in[4];
  const float* Wih0  = (const float*)d_in[5];
  const float* Whh0  = (const float*)d_in[6];
  const float* bih0  = (const float*)d_in[7];
  const float* bhh0  = (const float*)d_in[8];
  const float* Wih1  = (const float*)d_in[9];
  const float* Whh1  = (const float*)d_in[10];
  const float* bih1  = (const float*)d_in[11];
  const float* bhh1  = (const float*)d_in[12];
  const float* embW  = (const float*)d_in[13];
  const float* embB  = (const float*)d_in[14];
  const float* g1W   = (const float*)d_in[15];
  const float* g1as  = (const float*)d_in[16];
  const float* g1ad  = (const float*)d_in[17];
  const float* g1eW  = (const float*)d_in[18];
  const float* g1ae  = (const float*)d_in[19];
  const float* g1b   = (const float*)d_in[20];
  const float* g2W   = (const float*)d_in[21];
  const float* g2as  = (const float*)d_in[22];
  const float* g2ad  = (const float*)d_in[23];
  const float* g2eW  = (const float*)d_in[24];
  const float* g2ae  = (const float*)d_in[25];
  const float* g2b   = (const float*)d_in[26];
  const float* outW  = (const float*)d_in[27];
  const float* outB  = (const float*)d_in[28];
  float* y = (float*)d_out;

  size_t off = 0;
  auto WS = [&](size_t bytes) -> void* {
    void* p = (char*)d_ws + off;
    off += (bytes + 255) & ~size_t(255);
    return p;
  };
  float*    emb   = (float*)WS(32 * 4);
  float*    hs0   = (float*)WS((size_t)TT * 64 * 4);
  float*    q1    = (float*)WS(20 * 4);
  float*    q2    = (float*)WS(5 * 4);
  _Float16* BT1   = (_Float16*)WS((size_t)256 * 64 * 2);
  _Float16* BT2   = (_Float16*)WS((size_t)32 * 256 * 2);
  _Float16* Xin   = (_Float16*)WS((size_t)NP * 64 * 2);
  float*    H1    = (float*)WS((size_t)NP * 256 * 4);
  float*    deg   = (float*)WS((size_t)NN * 4);
  float*    meane = (float*)WS((size_t)NN * 5 * 4);
  float*    as1   = (float*)WS((size_t)NN * 4 * 4);
  float*    ad1   = (float*)WS((size_t)NN * 4 * 4);
  unsigned* amax1 = (unsigned*)WS((size_t)NN * 4 * 4);
  float*    den1  = (float*)WS((size_t)NN * 4 * 4);
  float*    alph1 = (float*)WS((size_t)EE2 * 4 * 4);
  float*    out1  = (float*)WS((size_t)NP * 256 * 4);
  _Float16* H1h   = (_Float16*)WS((size_t)NP * 256 * 2);
  float*    H2    = (float*)WS((size_t)NP * 32 * 4);
  float*    as2   = (float*)WS((size_t)NN * 4);
  float*    ad2   = (float*)WS((size_t)NN * 4);
  unsigned* amax2 = (unsigned*)WS((size_t)NN * 4);
  float*    den2  = (float*)WS((size_t)NN * 4);
  float*    alph2 = (float*)WS((size_t)EE2 * 4);
  float*    out2  = (float*)WS((size_t)NN * 32 * 4);

  auto G = [](int n) { return (n + 255) / 256; };

  /* zero accumulators (harness does not re-zero between replays) */
  k_zero<<<G(NN), 256, 0, stream>>>(deg, NN);
  k_zero<<<G(NN * 5), 256, 0, stream>>>(meane, NN * 5);
  k_zero<<<G(NN * 4), 256, 0, stream>>>((float*)amax1, NN * 4);
  k_zero<<<G(NN * 4), 256, 0, stream>>>(den1, NN * 4);
  k_zero<<<G(NP * 256), 256, 0, stream>>>(out1, NP * 256);
  k_zero<<<G(NN), 256, 0, stream>>>((float*)amax2, NN);
  k_zero<<<G(NN), 256, 0, stream>>>(den2, NN);
  k_zero<<<G(NN * 32), 256, 0, stream>>>(out2, NN * 32);

  k_lstm<<<1, 64, 0, stream>>>(env, Wih0, Whh0, bih0, bhh0, Wih1, Whh1, bih1, bhh1,
                               embW, embB, hs0, emb);
  k_prep_w<<<1, 256, 0, stream>>>(g1W, g2W, g1eW, g1ae, g2eW, g2ae, BT1, BT2, q1, q2);
  k_xin<<<G(NP), 256, 0, stream>>>(x, emb, Xin);
  k_deg<<<G(EEDG), 256, 0, stream>>>(ei, eattr, deg, meane);
  k_meandiv<<<G(NN * 5), 256, 0, stream>>>(deg, meane);

  /* GEMM1: [NP,64]f16 x [64,256] -> H1 [NP,256] */
  {
    int tiles = (NP / 16) * (256 / 16);
    k_gemm_wmma<<<(tiles + 3) / 4, 128, 0, stream>>>(Xin, BT1, H1, NP / 16, 16, 64, 256);
  }
  k_ad1<<<G(NN * 4), 256, 0, stream>>>(H1, g1as, g1ad, as1, ad1);
  k_alpha1<<<G(EE2), 256, 0, stream>>>(ei, eattr, meane, as1, ad1, q1, alph1, amax1);
  k_exp1<<<G(EE2 * 4), 256, 0, stream>>>(ei, amax1, alph1, den1);
  k_agg1<<<G(EE2 * 4), 256, 0, stream>>>(ei, alph1, den1, H1, out1);
  k_elu1<<<G(NP * 256), 256, 0, stream>>>(out1, g1b, H1h);

  /* GEMM2: [NP,256]f16 x [256,32] -> H2 [NP,32] */
  {
    int tiles = (NP / 16) * (32 / 16);
    k_gemm_wmma<<<(tiles + 3) / 4, 128, 0, stream>>>(H1h, BT2, H2, NP / 16, 2, 256, 32);
  }
  k_ad2<<<G(NN), 256, 0, stream>>>(H2, g2as, g2ad, as2, ad2);
  k_alpha2<<<G(EE2), 256, 0, stream>>>(ei, eattr, meane, as2, ad2, q2, alph2, amax2);
  k_exp2<<<G(EE2), 256, 0, stream>>>(ei, amax2, alph2, den2);
  k_agg2<<<G(EE2), 256, 0, stream>>>(ei, alph2, den2, H2, out2);
  k_final<<<G(NN), 256, 0, stream>>>(x, scmax, out2, g2b, outW, outB, y);
}
